// VINet_86809878986754
// MI455X (gfx1250) — compile-verified, matrix operand order checked by reading, not run on hardware
//
#include <hip/hip_runtime.h>
#include <hip/hip_bf16.h>

// ---------------------------------------------------------------------------
// CDNA5 (gfx1250) VINet forward.
//  - Convs: implicit GEMM, 64(M) x 128(N) macro-tile, K-step 32.
//      * A (weights): double-buffered TDM tensor_load_to_lds (OOB zero-fill),
//        s_wait_tensorcnt 1/0 pipeline.
//      * B (im2col): VALU gather (incremental k-decode) -> ds_store_b128.
//      * 8 waves (4M x 2N); each wave 16x64 -> 4 x v_wmma_f32_16x16x32_f16.
//  - rnn0 GEMV (805 MB weights): memory-bound fp32 streaming dot + prefetch.
//  - Correlation / tiny LSTMs / FC head: VALU kernels.
// ---------------------------------------------------------------------------

typedef __attribute__((ext_vector_type(16))) _Float16     v16h;
typedef __attribute__((ext_vector_type(8)))  _Float16     h8;
typedef __attribute__((ext_vector_type(8)))  float        v8f;
typedef __attribute__((ext_vector_type(4)))  unsigned int u32x4;
typedef __attribute__((ext_vector_type(8)))  int          i32x8;
typedef __attribute__((ext_vector_type(4)))  int          i32x4;

__device__ __forceinline__ float sigf(float x) { return 1.0f / (1.0f + __expf(-x)); }

// ---------------------------------------------------------------------------
// TDM: async 2D f32 tile load Global -> LDS.
//   tile = tile1 rows x tile0 cols, row stride = stride0 elements.
//   Hardware zero-fills reads beyond (td0, td1) tensor extents.
// Descriptor bit layout per CDNA5 ISA.
// ---------------------------------------------------------------------------
__device__ __forceinline__ void tdm_load_2d_f32(
    unsigned lds_off, const float* gaddr,
    unsigned td0, unsigned td1,
    unsigned tile0, unsigned tile1, unsigned stride0)
{
    const unsigned long long ga = (unsigned long long)gaddr;
    u32x4 g0;
    g0[0] = 1u;                                                // count=1 (user D#)
    g0[1] = lds_off;                                           // lds_addr [63:32]
    g0[2] = (unsigned)(ga & 0xffffffffull);                    // global_addr lo
    g0[3] = (unsigned)((ga >> 32) & 0x01ffffffull)             // global_addr hi
            | 0x80000000u;                                     // type=2 ("image")
    i32x8 g1;
    g1[0] = (int)(2u << 16);                                   // data_size=4B
    g1[1] = (int)((td0 & 0xffffu) << 16);                      // tensor_dim0 lo16
    g1[2] = (int)(((td0 >> 16) & 0xffffu) |
                  ((td1 & 0xffffu) << 16));                    // td0 hi | td1 lo
    g1[3] = (int)(((td1 >> 16) & 0xffffu) |
                  ((tile0 & 0xffffu) << 16));                  // td1 hi | tile_dim0
    g1[4] = (int)(tile1 & 0xffffu);                            // tile_dim1 (tile_dim2=0)
    g1[5] = (int)stride0;                                      // tensor_dim0_stride lo32
    g1[6] = 0;
    g1[7] = 0;
    i32x4 z = {0, 0, 0, 0};
#if __clang_major__ >= 23
    i32x8 z8 = {0, 0, 0, 0, 0, 0, 0, 0};
    __builtin_amdgcn_tensor_load_to_lds(g0, g1, z, z, z8, 0);
#else
    __builtin_amdgcn_tensor_load_to_lds(g0, g1, z, z, 0);
#endif
}

// ===========================================================================
// Implicit-GEMM convolution, NCHW in / OIHW weights / NCHW out.
// GEMM: M = Cout, N = Hout*Wout, K = Cin*KH*KW.
// Block: 256 thr = 8 waves, macro-tile 64(M) x 128(N), K-step 32.
//   - A slab (64x32 f32): double-buffered TDM into LDS.
//   - B slab (128 cols x 32 k, f16): im2col gather + ds_store_b128,
//     overlapped with the in-flight TDM of the NEXT A slab.
//   - waves 4(M) x 2(N); each wave: 1 A-frag, 4 B-frags, 4 WMMAs.
// ===========================================================================
__global__ __launch_bounds__(256) void conv_wmma_kernel(
    const float* __restrict__ in, const float* __restrict__ wgt,
    const float* __restrict__ bias, float* __restrict__ out,
    int Cin, int H, int W, int Cout, int KH, int KW,
    int stride, int pad, int Hout, int Wout, float slope)
{
    __shared__ float    Af[2][64][32];    // weight slabs (f32, TDM-filled)
    __shared__ _Float16 Bf[2][128][32];   // im2col slabs (f16), [col][k]

    const int Ntot = Hout * Wout;
    const int K    = Cin * KH * KW;
    const int KHW  = KH * KW;

    const int tid  = threadIdx.x;
    const int lane = tid & 31;
    const int wid  = tid >> 5;        // 0..7
    const int wm   = wid & 3;         // M sub-tile 0..3
    const int wn   = wid >> 2;        // N sub-tile 0..1
    const int m0   = blockIdx.y * 64;
    const int n0   = blockIdx.x * 128;

    const int l16 = lane & 15;
    const int hs  = lane >> 4;        // K-phase half-wave select

    const int mtile  = m0 + wm * 16;
    const int arow_l = wm * 16 + l16;

    const unsigned ldsA0 = (unsigned)(unsigned long long)(void*)&Af[0][0][0];
    const unsigned ldsA1 = (unsigned)(unsigned long long)(void*)&Af[1][0][0];

    v8f acc[4] = {};

    // prologue: slab 0 in flight
    if (wid == 0)
        tdm_load_2d_f32(ldsA0, wgt + (long)m0 * K,
                        (unsigned)K, (unsigned)(Cout - m0), 32u, 64u, (unsigned)K);

    int cb = 0;
    for (int kb = 0; kb < K; kb += 32, cb ^= 1) {
        __syncthreads();   // compute of iteration i-1 fully done

        const bool has_next = (kb + 32) < K;
        if (wid == 0 && has_next)
            tdm_load_2d_f32(cb ? ldsA0 : ldsA1, wgt + (long)m0 * K + kb + 32,
                            (unsigned)(K - kb - 32), (unsigned)(Cout - m0),
                            32u, 64u, (unsigned)K);

        // ---- B slab: 512 jobs (128 cols x 4 k-groups), 2 jobs/thread ----
#pragma unroll
        for (int jj = 0; jj < 2; ++jj) {
            const int job  = tid + jj * 256;
            const int bcol = job >> 2;
            const int bkg  = (job & 3) * 8;
            const int bn   = n0 + bcol;
            h8 t = {};
            if (bn < Ntot) {
                int oy = bn / Wout;
                int ox = bn - oy * Wout;
                const int iyb = oy * stride - pad;
                const int ixb = ox * stride - pad;
                const int k0  = kb + bkg;
                int ci = k0 / KHW;
                int r  = k0 - ci * KHW;
                int kh = r / KW;
                int kw = r - kh * KW;
#pragma unroll
                for (int j = 0; j < 8; ++j) {
                    float v = 0.0f;
                    if (k0 + j < K) {
                        const int iy = iyb + kh;
                        const int ix = ixb + kw;
                        if (iy >= 0 && iy < H && ix >= 0 && ix < W)
                            v = in[(long)ci * H * W + (long)iy * W + ix];
                    }
                    t[j] = (_Float16)v;
                    // incremental k -> (ci,kh,kw) advance (no divisions)
                    if (++kw == KW) { kw = 0; if (++kh == KH) { kh = 0; ++ci; } }
                }
            }
            *(h8*)&Bf[cb][bcol][bkg] = t;
        }

        // retire the CURRENT slab's TDM (oldest of <=2 outstanding, in-order)
        if (wid == 0) {
            if (has_next) __builtin_amdgcn_s_wait_tensorcnt(1);
            else          __builtin_amdgcn_s_wait_tensorcnt(0);
        }
        __syncthreads();   // slabs visible to all waves

        // ---- fragments (16-bit A 16x32 layout: lanes 0-15 K[0-7]+[16-23],
        //      lanes 16-31 K[8-15]+[24-31]) ----
        v16h a;
#pragma unroll
        for (int i = 0; i < 8; ++i) {
            a[i]     = (_Float16)Af[cb][arow_l][hs * 8 + i];
            a[i + 8] = (_Float16)Af[cb][arow_l][16 + hs * 8 + i];
        }
#pragma unroll
        for (int s = 0; s < 4; ++s) {
            const int col = wn * 64 + s * 16 + l16;
            v16h b;
#pragma unroll
            for (int i = 0; i < 8; ++i) {
                b[i]     = Bf[cb][col][hs * 8 + i];
                b[i + 8] = Bf[cb][col][16 + hs * 8 + i];
            }
            acc[s] = __builtin_amdgcn_wmma_f32_16x16x32_f16(
                false, a, false, b, (short)0, acc[s], false, false);
        }
    }

    // ---- epilogue: C/D layout lane%16 = N, VGPR r -> M = r + 8*(lane/16) ----
#pragma unroll
    for (int r = 0; r < 8; ++r) {
        const int m = mtile + r + 8 * hs;
        if (m < Cout) {
            const float bv = bias[m];
#pragma unroll
            for (int s = 0; s < 4; ++s) {
                const int nc = n0 + wn * 64 + s * 16 + l16;
                if (nc < Ntot) {
                    float v = acc[s][r] + bv;
                    v = (v > 0.0f) ? v : v * slope;   // slope==1 -> identity
                    out[(long)m * Ntot + nc] = v;
                }
            }
        }
    }
}

// ===========================================================================
// FlowNetC correlation: x1,x2 (256,64,96) -> out (441,64,96), mean over C,
// leaky relu. One thread per (displacement, pixel).
// ===========================================================================
__global__ __launch_bounds__(256) void correlate_kernel(
    const float* __restrict__ x1, const float* __restrict__ x2,
    float* __restrict__ out)
{
    const int HW  = 64 * 96;
    const int d   = blockIdx.y;
    const int pix = blockIdx.x * 256 + threadIdx.x;
    if (pix >= HW) return;
    const int oi = d / 21, oj = d - oi * 21;
    const int dy = oi * 2 - 20, dx = oj * 2 - 20;
    const int y  = pix / 96, xx = pix - y * 96;
    const int y2 = y + dy, x2c = xx + dx;
    float s = 0.0f;
    if (y2 >= 0 && y2 < 64 && x2c >= 0 && x2c < 96) {
        const float* p1 = x1 + pix;
        const float* p2 = x2 + y2 * 96 + x2c;
#pragma unroll 4
        for (int c = 0; c < 256; ++c)
            s += p1[c * HW] * p2[c * HW];
    }
    float v = s * (1.0f / 256.0f);
    v = (v > 0.0f) ? v : 0.01f * v;
    out[(long)d * HW + pix] = v;
}

// ===========================================================================
// GEMV: y[row] = act( W[row,:] . x + b1[row] + b2[row] )
// One block per row; streaming dot + LDS reduction; global_prefetch_b8.
// ===========================================================================
__global__ __launch_bounds__(256) void gemv_kernel(
    const float* __restrict__ Wm, const float* __restrict__ x,
    const float* __restrict__ b1, const float* __restrict__ b2,
    float* __restrict__ y, int M, int K, int act)
{
    __shared__ float red[256];
    const int row = blockIdx.x;
    if (row >= M) return;
    const float* wr = Wm + (long)row * (long)K;
    float s = 0.0f;
    for (int j = threadIdx.x; j < K; j += 256) {
        __builtin_prefetch(wr + j + 4096, 0, 1);
        s += wr[j] * x[j];
    }
    red[threadIdx.x] = s;
    __syncthreads();
#pragma unroll
    for (int off = 128; off > 0; off >>= 1) {
        if (threadIdx.x < off) red[threadIdx.x] += red[threadIdx.x + off];
        __syncthreads();
    }
    if (threadIdx.x == 0) {
        float v = red[0];
        if (b1) v += b1[row];
        if (b2) v += b2[row];
        if (act == 1) v = fmaxf(v, 0.0f);
        y[row] = v;
    }
}

// ===========================================================================
// LSTM cell, zero previous state, single step: h = sig(o)*tanh(sig(i)*tanh(g))
// ===========================================================================
__global__ void lstm_cell_zero_kernel(const float* __restrict__ g,
                                      float* __restrict__ h, int H)
{
    const int i = blockIdx.x * blockDim.x + threadIdx.x;
    if (i >= H) return;
    const float c = sigf(g[i]) * tanhf(g[2 * H + i]);
    h[i] = sigf(g[3 * H + i]) * tanhf(c);
}

// ===========================================================================
// Tiny 2-layer IMU LSTM (hidden=6, T=10), single wave.
// ===========================================================================
__global__ void imu_lstm_kernel(const float* __restrict__ imu,   // (10,6)
    const float* __restrict__ W0i, const float* __restrict__ W0h,
    const float* __restrict__ b0i, const float* __restrict__ b0h,
    const float* __restrict__ W1i, const float* __restrict__ W1h,
    const float* __restrict__ b1i, const float* __restrict__ b1h,
    float* __restrict__ out6)
{
    __shared__ float h[6], c[6], y0[10][6], gates[24];
    const int t = threadIdx.x;

    if (t < 6) { h[t] = 0.0f; c[t] = 0.0f; }
    __syncthreads();
    for (int step = 0; step < 10; ++step) {
        if (t < 24) {
            float g = b0i[t] + b0h[t];
            for (int k = 0; k < 6; ++k)
                g += W0i[t * 6 + k] * imu[step * 6 + k] + W0h[t * 6 + k] * h[k];
            gates[t] = g;
        }
        __syncthreads();
        if (t < 6) {
            float cn = sigf(gates[6 + t]) * c[t] + sigf(gates[t]) * tanhf(gates[12 + t]);
            c[t] = cn;
            h[t] = sigf(gates[18 + t]) * tanhf(cn);
            y0[step][t] = h[t];
        }
        __syncthreads();
    }
    if (t < 6) { h[t] = 0.0f; c[t] = 0.0f; }
    __syncthreads();
    for (int step = 0; step < 10; ++step) {
        if (t < 24) {
            float g = b1i[t] + b1h[t];
            for (int k = 0; k < 6; ++k)
                g += W1i[t * 6 + k] * y0[step][k] + W1h[t * 6 + k] * h[k];
            gates[t] = g;
        }
        __syncthreads();
        if (t < 6) {
            float cn = sigf(gates[6 + t]) * c[t] + sigf(gates[t]) * tanhf(gates[12 + t]);
            c[t] = cn;
            h[t] = sigf(gates[18 + t]) * tanhf(cn);
        }
        __syncthreads();
    }
    if (t < 6) out6[t] = h[t];
}

// ===========================================================================
// Concatenate [feat(98304), imu_out(6), xyzq(7)] -> xin(98317)
// ===========================================================================
__global__ void assemble_kernel(const float* __restrict__ feat,
                                const float* __restrict__ imu6,
                                const float* __restrict__ xyzq,
                                float* __restrict__ xin)
{
    const int i = blockIdx.x * 256 + threadIdx.x;
    if (i < 98304)       xin[i] = feat[i];
    else if (i < 98310)  xin[i] = imu6[i - 98304];
    else if (i < 98317)  xin[i] = xyzq[i - 98310];
}

// ===========================================================================
// Host-side orchestration
// ===========================================================================
extern "C" void kernel_launch(void* const* d_in, const int* in_sizes, int n_in,
                              void* d_out, int out_size, void* d_ws, size_t ws_size,
                              hipStream_t stream)
{
    (void)in_sizes; (void)n_in; (void)out_size; (void)ws_size;
#define P(n) ((const float*)d_in[(n)])
    const float* x    = P(0);     // (1,2,512,768)
    const float* imu  = P(1);     // (1,10,6)
    const float* xyzq = P(2);     // (1,1,7)
    const float *conv1_w = P(3),  *conv1_b = P(4);
    const float *conv2_w = P(5),  *conv2_b = P(6);
    const float *conv3_w = P(7),  *conv3_b = P(8);
    const float *redir_w = P(9),  *redir_b = P(10);
    const float *conv31_w = P(11), *conv31_b = P(12);
    const float *conv4_w = P(13), *conv4_b = P(14);
    const float *conv41_w = P(15), *conv41_b = P(16);
    const float *conv5_w = P(17), *conv5_b = P(18);
    const float *conv51_w = P(19), *conv51_b = P(20);
    const float *conv6_w = P(21), *conv6_b = P(22);
    const float *conv61_w = P(23), *conv61_b = P(24);
    const float *imu0_Wih = P(25), *imu0_Whh = P(26), *imu0_bih = P(27), *imu0_bhh = P(28);
    const float *imu1_Wih = P(29), *imu1_Whh = P(30), *imu1_bih = P(31), *imu1_bhh = P(32);
    const float *rnn0_Wih = P(33); /* P(34)=rnn0_Whh unused: h0 == 0 */
    const float *rnn0_bih = P(35), *rnn0_bhh = P(36);
    const float *rnn1_Wih = P(37); /* P(38)=rnn1_Whh unused: h0 == 0 */
    const float *rnn1_bih = P(39), *rnn1_bhh = P(40);
    const float *fc1_w = P(41), *fc1_b = P(42);
    const float *fc2_w = P(43), *fc2_b = P(44);
    const float *fco_w = P(45), *fco_b = P(46);
#undef P

    // ---- workspace layout (floats) ----
    float* Wf = (float*)d_ws;
    size_t off = 0;
    auto alloc = [&](size_t n) { float* p = Wf + off; off += n; return p; };
    float* bufA   = alloc(6291456);   // 64 x 256 x 384
    float* bufB   = alloc(3145728);   // 128 x 128 x 192
    float* x1f    = alloc(1572864);   // 256 x 64 x 96
    float* x2f    = alloc(1572864);
    float* catb   = alloc(2906112);   // 473 x 64 x 96
    float* bufC   = alloc(1572864);
    float* bufD   = alloc(786432);
    float* xin    = alloc(98320);
    float* gates  = alloc(2048);
    float* h1     = alloc(512);
    float* gates2 = alloc(2048);
    float* h2     = alloc(512);
    float* f1     = alloc(128);
    float* f2     = alloc(32);
    float* imu_o  = alloc(8);

    auto conv = [&](const float* in, const float* w, const float* b, float* out,
                    int Cin, int H, int W, int Cout, int KH, int KW,
                    int s, int p, float slope) {
        const int Ho = (H + 2 * p - KH) / s + 1;
        const int Wo = (W + 2 * p - KW) / s + 1;
        dim3 grid((Ho * Wo + 127) / 128, (Cout + 63) / 64);
        conv_wmma_kernel<<<grid, 256, 0, stream>>>(in, w, b, out,
            Cin, H, W, Cout, KH, KW, s, p, Ho, Wo, slope);
    };
    auto gemv = [&](const float* Wm, const float* xv, const float* b1,
                    const float* b2, float* y, int M, int K, int act) {
        gemv_kernel<<<M, 256, 0, stream>>>(Wm, xv, b1, b2, y, M, K, act);
    };

    const float LR = 0.01f;

    // ---- encoder, image 0 ----
    conv(x,           conv1_w, conv1_b, bufA, 1,   512, 768, 64,  7, 7, 2, 3, LR);
    conv(bufA,        conv2_w, conv2_b, bufB, 64,  256, 384, 128, 5, 5, 2, 2, LR);
    conv(bufB,        conv3_w, conv3_b, x1f,  128, 128, 192, 256, 5, 5, 2, 2, LR);
    // ---- encoder, image 1 ----
    conv(x + 512*768, conv1_w, conv1_b, bufA, 1,   512, 768, 64,  7, 7, 2, 3, LR);
    conv(bufA,        conv2_w, conv2_b, bufB, 64,  256, 384, 128, 5, 5, 2, 2, LR);
    conv(bufB,        conv3_w, conv3_b, x2f,  128, 128, 192, 256, 5, 5, 2, 2, LR);

    // ---- redir (cat channels 0..31) + correlation (channels 32..472) ----
    conv(x1f, redir_w, redir_b, catb, 256, 64, 96, 32, 1, 1, 1, 0, LR);
    {
        dim3 grid((64 * 96 + 255) / 256, 441);
        correlate_kernel<<<grid, 256, 0, stream>>>(x1f, x2f, catb + 32 * 64 * 96);
    }

    // ---- refinement stack ----
    conv(catb, conv31_w, conv31_b, bufC, 473,  64, 96, 256,  3, 3, 1, 1, LR);
    conv(bufC, conv4_w,  conv4_b,  bufD, 256,  64, 96, 512,  3, 3, 2, 1, LR);
    conv(bufD, conv41_w, conv41_b, bufC, 512,  32, 48, 512,  3, 3, 1, 1, LR);
    conv(bufC, conv5_w,  conv5_b,  bufD, 512,  32, 48, 512,  3, 3, 2, 1, LR);
    conv(bufD, conv51_w, conv51_b, bufC, 512,  16, 24, 512,  3, 3, 1, 1, LR);
    conv(bufC, conv6_w,  conv6_b,  bufD, 512,  16, 24, 1024, 3, 3, 2, 1, LR);
    conv(bufD, conv61_w, conv61_b, bufC, 1024, 8,  12, 1024, 3, 3, 1, 1, 1.0f);

    // ---- IMU LSTM ----
    imu_lstm_kernel<<<1, 32, 0, stream>>>(imu,
        imu0_Wih, imu0_Whh, imu0_bih, imu0_bhh,
        imu1_Wih, imu1_Whh, imu1_bih, imu1_bhh, imu_o);

    // ---- assemble xin = [feat, imu_out, xyzq] ----
    assemble_kernel<<<(98317 + 255) / 256, 256, 0, stream>>>(bufC, imu_o, xyzq, xin);

    // ---- rnn0 / rnn1 (single step, zero initial state) ----
    gemv(rnn0_Wih, xin, rnn0_bih, rnn0_bhh, gates,  2048, 98317, 0);
    lstm_cell_zero_kernel<<<2, 256, 0, stream>>>(gates,  h1, 512);
    gemv(rnn1_Wih, h1,  rnn1_bih, rnn1_bhh, gates2, 2048, 512,   0);
    lstm_cell_zero_kernel<<<2, 256, 0, stream>>>(gates2, h2, 512);

    // ---- FC head ----
    gemv(fc1_w, h2, fc1_b, nullptr, f1, 128, 512, 1);
    gemv(fc2_w, f1, fc2_b, nullptr, f2, 32,  128, 1);
    gemv(fco_w, f2, fco_b, nullptr, (float*)d_out, 6, 32, 0);
}
